// FourierCrossAttention_3461743640800
// MI455X (gfx1250) — compile-verified, matrix operand order checked by reading, not run on hardware
//
#include <hip/hip_runtime.h>
#include <hip/hip_bf16.h>
#include <math.h>

// ---------------------------------------------------------------------------
// FourierCrossAttention for MI455X (gfx1250, wave32, WMMA).
//
// Sizes: B=8, L=8192, H=8, E=64, MODES=64, K=64, HE=H*E=512.
// Heavy stage = 64-mode direct DFT of q and k == f16 WMMA GEMM:
//   [128 (Re/Im x modes)] x [8192 (time)] x [512 (h*e)] per (tensor, batch).
// Memory floor: 268MB @ 23.3TB/s ~ 11.5us; f16 WMMA rate ~ matches -> the
// GEMM must run on matrix cores with a single HBM pass over q and k.
// A-panel (f16 twiddles, L2-resident) is double-buffered into LDS with
// GLOBAL_LOAD_ASYNC_TO_LDS_B128 (ASYNCcnt); B-panel is staged f32->f16 via
// VGPRs (conversion required) with one b128 load per thread per chunk.
// ---------------------------------------------------------------------------

typedef __attribute__((ext_vector_type(16))) _Float16 v16h;
typedef __attribute__((ext_vector_type(8)))  _Float16 v8h;
typedef __attribute__((ext_vector_type(8)))  float    v8f;
typedef __attribute__((ext_vector_type(4)))  float    v4f;

#define L_SEQ   8192
#define HE      512
#define MROWS   128      // 64 cos rows + 64 (-sin) rows
#define NB      8        // batches
#define NH      8
#define NE      64
#define NK      64
#define NMODE   64
#define TWO_PI  6.28318530717958647692f

// LDS layout (halves): A double buffer [2][128][64], then Bs[16][72].
#define A_CHUNK_HALVES  (MROWS * 64)          // 8192 halves = 16KB
#define BS_OFF_HALVES   (2 * A_CHUNK_HALVES)  // 32KB in
#define BS_STRIDE       72                    // 144B rows: 16B aligned, 9-bank skew

// ---------------------------------------------------------------------------
// Kernel 1: per-tensor f16 DFT basis.  Wt layout: [tensor][row 0..127][t].
//   row r < 64 :  cos(2*pi*mode*t/L)   (Re of e^{-i*2*pi*m*t/L})
//   row r >= 64: -sin(2*pi*mode*t/L)   (Im)
// ---------------------------------------------------------------------------
__global__ __launch_bounds__(256) void twiddle_kernel(
    const int* __restrict__ idx_q, const int* __restrict__ idx_kv,
    _Float16* __restrict__ Wt) {
  int gid = blockIdx.x * 256 + threadIdx.x;          // 2*128*8192 total
  int t    = gid & (L_SEQ - 1);
  int r    = (gid >> 13) & (MROWS - 1);
  int tsel = gid >> 20;
  const int* sel = tsel ? idx_kv : idx_q;
  int mode = sel[r & (NMODE - 1)];
  int ph = (mode * t) & (L_SEQ - 1);                 // exact phase reduction
  float th = TWO_PI * (float)ph * (1.0f / (float)L_SEQ);
  float v = (r < NMODE) ? cosf(th) : -sinf(th);
  Wt[gid] = (_Float16)v;
}

// ---------------------------------------------------------------------------
// Kernel 2: DFT GEMM with v_wmma_f32_16x16x32_f16 + async-to-LDS A panel.
// grid = (32 column-blocks of 16, 8 batches, 2 tensors); 256 thr = 8 waves.
// Wave w computes rows [16w, 16w+16) of the 128-row output, 16 columns.
// ---------------------------------------------------------------------------
__device__ __forceinline__ void async_copy_b128(unsigned lds_off,
                                                const _Float16* gsrc) {
  // GLOBAL_LOAD_ASYNC_TO_LDS_B128 (ISA 15.18.3, op 98): vdst = LDS byte
  // offset in a VGPR, vaddr = 64-bit global address. Tracked by ASYNCcnt.
  asm volatile("global_load_async_to_lds_b128 %0, %1, off"
               :
               : "v"(lds_off), "v"(gsrc)
               : "memory");
}

__global__ __launch_bounds__(256) void dft_wmma_kernel(
    const float* __restrict__ q, const float* __restrict__ k,
    const _Float16* __restrict__ Wt, float* __restrict__ X) {
  const int c0   = blockIdx.x * 16;                  // column block (h*e)
  const int b    = blockIdx.y;
  const int tsel = blockIdx.z;                       // 0 = q, 1 = k
  const float* __restrict__ xb =
      (tsel == 0 ? q : k) + (size_t)b * L_SEQ * HE;
  const _Float16* __restrict__ Wbase =
      Wt + (size_t)tsel * MROWS * L_SEQ;

  const int tid  = threadIdx.x;
  const int lane = tid & 31;
  const int w    = tid >> 5;                         // wave 0..7
  const int m    = lane & 15;
  const int grp  = lane >> 4;                        // 0 or 1

  __shared__ __align__(16) _Float16 lds[2 * A_CHUNK_HALVES + 16 * BS_STRIDE];
  _Float16* Bs = lds + BS_OFF_HALVES;
  // Low 32 bits of the generic (flat) address of a __shared__ object are the
  // LDS byte offset (ISA 10.2).
  const unsigned lds_base = (unsigned)(uintptr_t)(void*)lds;

  // Async A-panel copy: chunk = rows 0..127, cols [kk, kk+64) of Wt[tsel],
  // packed row-major [128][64] halves (128B rows -> 8 x 16B segments/row).
  // 1024 segments, 256 threads -> 4 async b128 per thread per chunk.
  const int seg_row = tid >> 1;                      // with p: (tid+256p)>>3
  (void)seg_row;

  v8f acc = {};

  // --- prologue: issue chunk 0 into buffer 0 ---
#pragma unroll
  for (int p = 0; p < 4; ++p) {
    int seg = tid + p * 256;                         // 0..1023
    int row = seg >> 3;                              // 0..127
    int cs  = seg & 7;                               // 16B segment in row
    const _Float16* g = Wbase + (size_t)row * L_SEQ + 0 + cs * 8;
    unsigned l = lds_base + (unsigned)((row * 64 + cs * 8) * 2);
    async_copy_b128(l, g);
  }

  int buf = 0;
  for (int kk = 0; kk < L_SEQ; kk += 64) {
    // --- stage B: 64 t-rows x 16 cols of x, f32 -> f16, transposed.
    //     One b128 load per thread (row = tid/4, 4 cols each). ---
    {
      int row = tid >> 2;                            // 0..63
      int col = (tid & 3) * 4;                       // 0,4,8,12
      v4f v = *(const v4f*)(xb + (size_t)(kk + row) * HE + c0 + col);
      Bs[(col + 0) * BS_STRIDE + row] = (_Float16)v.x;
      Bs[(col + 1) * BS_STRIDE + row] = (_Float16)v.y;
      Bs[(col + 2) * BS_STRIDE + row] = (_Float16)v.z;
      Bs[(col + 3) * BS_STRIDE + row] = (_Float16)v.w;
    }

    // --- issue async A copy for the NEXT chunk, retire the current one ---
    if (kk + 64 < L_SEQ) {
      const int nbuf = buf ^ 1;
#pragma unroll
      for (int p = 0; p < 4; ++p) {
        int seg = tid + p * 256;
        int row = seg >> 3;
        int cs  = seg & 7;
        const _Float16* g = Wbase + (size_t)row * L_SEQ + (kk + 64) + cs * 8;
        unsigned l = lds_base +
                     (unsigned)((nbuf * A_CHUNK_HALVES + row * 64 + cs * 8) * 2);
        async_copy_b128(l, g);
      }
      __builtin_prefetch(xb + (size_t)(kk + 64 + (tid >> 2)) * HE + c0, 0, 1);
      // 4 outstanding from current chunk complete in-order first.
      asm volatile("s_wait_asynccnt 4" ::: "memory");
    } else {
      asm volatile("s_wait_asynccnt 0" ::: "memory");
    }
    __syncthreads();   // all waves' async copies + B stores visible

    const _Float16* As = lds + buf * A_CHUNK_HALVES;
#pragma unroll
    for (int s = 0; s < 2; ++s) {
      // A fragment (16x32 f16): lane -> M = lane%16, element i ->
      // K = (i<8?0:16) + 8*(lane/16) + i%8  => two contiguous 16B ds loads.
      const _Float16* arow = As + (w * 16 + m) * 64 + s * 32 + grp * 8;
      v8h alo = *(const v8h*)(arow);
      v8h ahi = *(const v8h*)(arow + 16);
      v16h a = __builtin_shufflevector(alo, ahi, 0, 1, 2, 3, 4, 5, 6, 7, 8, 9,
                                       10, 11, 12, 13, 14, 15);
      // B fragment (32x16 f16): mirrored indexing.
      const _Float16* brow = Bs + m * BS_STRIDE + s * 32 + grp * 8;
      v8h blo = *(const v8h*)(brow);
      v8h bhi = *(const v8h*)(brow + 16);
      v16h bf = __builtin_shufflevector(blo, bhi, 0, 1, 2, 3, 4, 5, 6, 7, 8, 9,
                                        10, 11, 12, 13, 14, 15);
      acc = __builtin_amdgcn_wmma_f32_16x16x32_f16(
          /*neg_a=*/false, a, /*neg_b=*/false, bf,
          /*c_mod=*/(short)0, acc, /*reuse_a=*/false, /*reuse_b=*/false);
    }
    __syncthreads();   // protect Bs / A buffer reuse
    buf ^= 1;
  }

  // C/D layout: VGPR r holds M = r + 8*(lane/16), N = lane%16.
  float* Xout = X + (size_t)(tsel * NB + b) * MROWS * HE;
#pragma unroll
  for (int r = 0; r < 8; ++r) {
    int row = w * 16 + grp * 8 + r;
    Xout[(size_t)row * HE + c0 + m] = acc[r];
  }
}

// ---------------------------------------------------------------------------
// Kernel 3: per-(b,h) tail. All matrices are 64x64 complex.
//   xqk[m][l] = sum_e Xq[l]*Xk[m]          (complex, no conj)
//   xqk = tanh(Re) + i*tanh(Im)
//   T[l][e]  = sum_m xqk[m][l]*Xk[m][e]
//   Y[l][kk] = sum_e T[l][e]*(wre + i*wim)[l,h,e,kk]
// ---------------------------------------------------------------------------
__global__ __launch_bounds__(256) void mix_kernel(
    const float* __restrict__ X, const float* __restrict__ wre,
    const float* __restrict__ wim, float* __restrict__ Yre,
    float* __restrict__ Yim) {
  const int b = blockIdx.x >> 3;
  const int h = blockIdx.x & 7;
  const int tid = threadIdx.x;

  __shared__ float qre[64][64], qim[64][64];         // Xq, later xqk
  __shared__ float kre[64][64], kim[64][64];         // Xk, later T

  const size_t bq = (size_t)(0 * NB + b) * MROWS * HE + h * NE;
  const size_t bk = (size_t)(1 * NB + b) * MROWS * HE + h * NE;
  for (int j = tid; j < 64 * 64; j += 256) {
    int mm = j >> 6, e = j & 63;
    qre[mm][e] = X[bq + (size_t)mm * HE + e];
    qim[mm][e] = X[bq + (size_t)(mm + 64) * HE + e];
    kre[mm][e] = X[bk + (size_t)mm * HE + e];
    kim[mm][e] = X[bk + (size_t)(mm + 64) * HE + e];
  }
  __syncthreads();

  float rr[16], ri[16];
  // scores + tanh
#pragma unroll
  for (int jj = 0; jj < 16; ++jj) {
    int flat = tid + jj * 256;
    int mm = flat >> 6, l = flat & 63;
    float ar = 0.f, ai = 0.f;
    for (int e = 0; e < 64; ++e) {
      float qr = qre[l][e], qi = qim[l][e];
      float kr = kre[mm][e], ki = kim[mm][e];
      ar += qr * kr - qi * ki;
      ai += qr * ki + qi * kr;
    }
    rr[jj] = tanhf(ar);
    ri[jj] = tanhf(ai);
  }
  __syncthreads();
#pragma unroll
  for (int jj = 0; jj < 16; ++jj) {
    int flat = tid + jj * 256;
    int mm = flat >> 6, l = flat & 63;
    qre[mm][l] = rr[jj];
    qim[mm][l] = ri[jj];
  }
  __syncthreads();

  // T = xqk^T(m,l) * Xk(m,e)
#pragma unroll
  for (int jj = 0; jj < 16; ++jj) {
    int flat = tid + jj * 256;
    int l = flat >> 6, e = flat & 63;
    float tr = 0.f, ti = 0.f;
    for (int mm = 0; mm < 64; ++mm) {
      float sr = qre[mm][l], si = qim[mm][l];
      float kr = kre[mm][e], ki = kim[mm][e];
      tr += sr * kr - si * ki;
      ti += sr * ki + si * kr;
    }
    rr[jj] = tr;
    ri[jj] = ti;
  }
  __syncthreads();
#pragma unroll
  for (int jj = 0; jj < 16; ++jj) {
    int flat = tid + jj * 256;
    int l = flat >> 6, e = flat & 63;
    kre[l][e] = rr[jj];
    kim[l][e] = ri[jj];
  }
  __syncthreads();

  // contract with complex weights w[l,h,e,kk]
#pragma unroll
  for (int jj = 0; jj < 16; ++jj) {
    int flat = tid + jj * 256;
    int l = flat >> 6, kk = flat & 63;
    float re = 0.f, im = 0.f;
    const size_t wb = ((size_t)(l * NH + h) * NE) * NK + kk;
    for (int e = 0; e < 64; ++e) {
      float tr = kre[l][e], ti = kim[l][e];
      float wr = wre[wb + (size_t)e * NK];
      float wi = wim[wb + (size_t)e * NK];
      re += tr * wr - ti * wi;
      im += tr * wi + ti * wr;
    }
    size_t o = ((size_t)(b * 64 + l) * NH + h) * NK + kk;
    Yre[o] = re;
    Yim[o] = im;
  }
}

// ---------------------------------------------------------------------------
// Kernel 4: length-64 irfft over bins 0..32, scaled by 1/(CIN*COUT*n).
// out[b, t, h, kk], t in [0,64).
// ---------------------------------------------------------------------------
__global__ __launch_bounds__(256) void irfft_kernel(
    const float* __restrict__ Yre, const float* __restrict__ Yim,
    float* __restrict__ out) {
  int idx = blockIdx.x * 256 + threadIdx.x;          // 262144 total
  int kk = idx & 63;
  int h  = (idx >> 6) & 7;
  int t  = (idx >> 9) & 63;
  int b  = idx >> 15;
  const float scale = 1.0f / (512.0f * 512.0f * 64.0f);

  size_t base = ((size_t)(b * 64) * NH + h) * NK + kk;
  size_t fstride = (size_t)NH * NK;                  // step in freq dim
  float acc = Yre[base];                             // f = 0 (imag ignored)
  float re32 = Yre[base + 32 * fstride];             // f = n/2, cos(pi*t)
  acc += (t & 1) ? -re32 : re32;
  for (int f = 1; f < 32; ++f) {
    float th = TWO_PI * (float)((f * t) & 63) * (1.0f / 64.0f);
    float c = cosf(th), s = sinf(th);
    float re = Yre[base + (size_t)f * fstride];
    float im = Yim[base + (size_t)f * fstride];
    acc += 2.0f * (re * c - im * s);
  }
  out[idx] = acc * scale;
}

// ---------------------------------------------------------------------------
// Launch. Workspace map (bytes):
//   [0,   4MB) f16 twiddle tables  [2][128][8192]
//   [4MB, 8MB) X  f32              [2][8][128][512]
//   [8MB, 9MB) Yre f32             [8][64][8][64]
//   [9MB,10MB) Yim f32
// ---------------------------------------------------------------------------
extern "C" void kernel_launch(void* const* d_in, const int* in_sizes, int n_in,
                              void* d_out, int out_size, void* d_ws,
                              size_t ws_size, hipStream_t stream) {
  const float* q   = (const float*)d_in[0];
  const float* k   = (const float*)d_in[1];
  // d_in[2] = v (unused by reference), d_in[3] = mask (unused)
  const float* wre = (const float*)d_in[4];
  const float* wim = (const float*)d_in[5];
  const int* idx_q  = (const int*)d_in[6];
  const int* idx_kv = (const int*)d_in[7];
  float* out = (float*)d_out;

  char* ws = (char*)d_ws;
  _Float16* Wt = (_Float16*)(ws);
  float* X     = (float*)(ws + (4u << 20));
  float* Yre   = (float*)(ws + (8u << 20));
  float* Yim   = (float*)(ws + (9u << 20));

  twiddle_kernel<<<(2 * MROWS * L_SEQ) / 256, 256, 0, stream>>>(idx_q, idx_kv,
                                                                Wt);
  dft_wmma_kernel<<<dim3(HE / 16, NB, 2), 256, 0, stream>>>(q, k, Wt, X);
  mix_kernel<<<NB * NH, 256, 0, stream>>>(X, wre, wim, Yre, Yim);
  irfft_kernel<<<(NB * 64 * NH * NK) / 256, 256, 0, stream>>>(Yre, Yim, out);
}
